// BDHGraphModel_36636071035464
// MI455X (gfx1250) — matrix operand order repaired and checked by reading, not compiled, not run on hardware
//
#include <hip/hip_runtime.h>

// ---------------------------------------------------------------------------
// BDH graph model for MI455X (gfx1250, wave32, WMMA).
//   - Sequential edge-recurrence (tiny, latency-bound): chain of small kernels
//     per timestep with global atomics (state fits in L2).
//   - Deferred batched readout GEMM [512 x 32000 x 2048] with
//     v_wmma_f32_16x16x32_bf16. W_ro is pre-converted to bf16 once per launch
//     (131 MB, L2-resident) so the GEMM hot loop is loads+WMMA only.
// ---------------------------------------------------------------------------

#define B_ 8
#define T_ 64
#define N_ 2048
#define E_ 65536
#define V_ 32000
#define K_ 2048
#define DECAY_ 0.99f

typedef __attribute__((ext_vector_type(16))) __bf16 v16bf;
typedef __attribute__((ext_vector_type(8)))  __bf16 v8bf;
typedef __attribute__((ext_vector_type(8)))  float  v8f;

__device__ __forceinline__ unsigned short bf16bits(float f) {
    unsigned u = __builtin_bit_cast(unsigned, f);
    u += 0x7fffu + ((u >> 16) & 1u);            // round-to-nearest-even
    return (unsigned short)(u >> 16);
}

__device__ __forceinline__ __bf16 f2bf(float f) {
    unsigned short h = bf16bits(f);
    return __builtin_bit_cast(__bf16, h);
}

__device__ __forceinline__ unsigned pack2(float a, float b) {
    return (unsigned)bf16bits(a) | ((unsigned)bf16bits(b) << 16);
}

__device__ __forceinline__ void cvt8(v16bf& d, int base, float4 a, float4 b) {
    d[base + 0] = f2bf(a.x); d[base + 1] = f2bf(a.y);
    d[base + 2] = f2bf(a.z); d[base + 3] = f2bf(a.w);
    d[base + 4] = f2bf(b.x); d[base + 5] = f2bf(b.y);
    d[base + 6] = f2bf(b.z); d[base + 7] = f2bf(b.w);
}

// ---------------------------------------------------------------- utilities
__global__ void zero_f(float* __restrict__ p, int n) {
    int i = blockIdx.x * blockDim.x + threadIdx.x;
    if (i < n) p[i] = 0.0f;
}

__global__ void copy_f(float* __restrict__ dst, const float* __restrict__ src, int n) {
    int i = blockIdx.x * blockDim.x + threadIdx.x;
    if (i < n) dst[i] = src[i];
}

__global__ void zero3(float* __restrict__ a, float* __restrict__ b,
                      float* __restrict__ c, int n) {
    int i = blockIdx.x * blockDim.x + threadIdx.x;
    if (i < n) { a[i] = 0.0f; b[i] = 0.0f; c[i] = 0.0f; }
}

// Streaming fp32 -> bf16, 8 elements per thread (2x float4 in, 1x uint4 out).
__global__ void cvt_f32_bf16_vec8(const float* __restrict__ src,
                                  uint4* __restrict__ dst, int nvec) {
    int i = blockIdx.x * blockDim.x + threadIdx.x;
    if (i >= nvec) return;
    const float4* s = reinterpret_cast<const float4*>(src) + (size_t)i * 2;
    float4 a = s[0], b = s[1];
    uint4 o;
    o.x = pack2(a.x, a.y); o.y = pack2(a.z, a.w);
    o.z = pack2(b.x, b.y); o.w = pack2(b.z, b.w);
    dst[i] = o;
}

// x[b][n] = embedding[idx[b][t]][n];  y_prev = 0
__global__ void gather_x(const float* __restrict__ emb, const int* __restrict__ idx,
                         int t, float* __restrict__ x, float* __restrict__ y) {
    int i = blockIdx.x * blockDim.x + threadIdx.x;    // [0, B_*N_)
    if (i >= B_ * N_) return;
    int b = i >> 11;            // /N_
    int n = i & (N_ - 1);
    int row = idx[b * T_ + t];
    x[i] = emb[(size_t)row * N_ + n];
    y[i] = 0.0f;
}

// A[:,dst] += x[:,src]*sigma ; hebbian = mean_b(y_prev[:,src]*x[:,dst]) ;
// sigma = (sigma + hebbian*Gs)*DECAY
__global__ void edge_pass1(const int* __restrict__ srcI, const int* __restrict__ dstI,
                           const float* __restrict__ Gs, float* __restrict__ sigma,
                           const float* __restrict__ x, const float* __restrict__ y,
                           float* __restrict__ A) {
    int e = blockIdx.x * blockDim.x + threadIdx.x;
    if (e >= E_) return;
    int s = srcI[e], d = dstI[e];
    float sg = sigma[e];
    float hebb = 0.0f;
#pragma unroll
    for (int b = 0; b < B_; ++b) {
        atomicAdd(&A[b * N_ + d], x[b * N_ + s] * sg);
        hebb += y[b * N_ + s] * x[b * N_ + d];
    }
    hebb *= (1.0f / (float)B_);
    sigma[e] = (sg + hebb * Gs[e]) * DECAY_;
}

// y_new[:,dst] += relu(A[:,src]) * Gy
__global__ void edge_pass2(const int* __restrict__ srcI, const int* __restrict__ dstI,
                           const float* __restrict__ Gy, const float* __restrict__ A,
                           float* __restrict__ ynew) {
    int e = blockIdx.x * blockDim.x + threadIdx.x;
    if (e >= E_) return;
    int s = srcI[e], d = dstI[e];
    float g = Gy[e];
#pragma unroll
    for (int b = 0; b < B_; ++b)
        atomicAdd(&ynew[b * N_ + d], fmaxf(A[b * N_ + s], 0.0f) * g);
}

// x_acc[:,dst] += y_new[:,src] * Gx
__global__ void edge_pass3(const int* __restrict__ srcI, const int* __restrict__ dstI,
                           const float* __restrict__ Gx, const float* __restrict__ ynew,
                           float* __restrict__ xacc) {
    int e = blockIdx.x * blockDim.x + threadIdx.x;
    if (e >= E_) return;
    int s = srcI[e], d = dstI[e];
    float g = Gx[e];
#pragma unroll
    for (int b = 0; b < B_; ++b)
        atomicAdd(&xacc[b * N_ + d], ynew[b * N_ + s] * g);
}

// x = relu(x_acc); y_prev = y_new; Xbf[b*T+t][n] = bf16(x) (layer-1 write wins)
__global__ void finalize_layer(const float* __restrict__ ynew, const float* __restrict__ xacc,
                               float* __restrict__ x, float* __restrict__ y,
                               unsigned short* __restrict__ Xbf, int t) {
    int i = blockIdx.x * blockDim.x + threadIdx.x;    // [0, B_*N_)
    if (i >= B_ * N_) return;
    int b = i >> 11;
    int n = i & (N_ - 1);
    float xv = fmaxf(xacc[i], 0.0f);
    x[i] = xv;
    y[i] = ynew[i];
    int m = b * T_ + t;                 // logits layout [B, T, V] -> row m
    Xbf[(size_t)m * K_ + n] = bf16bits(xv);
}

// ------------------------------------------------------------- readout GEMM
// out[m, col] = sum_k X[m,k] * W[col,k] + bias[col]
// M = 512 (=B*T), Ncols = 32000, K = 2048. bf16 WMMA, f32 accumulate.
// One wave: 1 N-tile (16 cols) x 4 M-tiles (64 rows). Block = 8 waves.

__device__ __forceinline__ v16bf load_afrag(const __bf16* __restrict__ Xbf,
                                            int row, int k, int hi) {
    // A fragment (16x32 bf16): lane holds row, elements 0..15 at K = hi*16+0..15
    // contiguous -> two 16B loads, register concat.
    const __bf16* arow = Xbf + (size_t)row * K_ + k;
    v8bf lo = *reinterpret_cast<const v8bf*>(arow + hi * 16);
    v8bf hh = *reinterpret_cast<const v8bf*>(arow + hi * 16 + 8);
    return __builtin_shufflevector(lo, hh, 0, 1, 2, 3, 4, 5, 6, 7,
                                           8, 9, 10, 11, 12, 13, 14, 15);
}

// Fast path: both operands pre-converted bf16. Hot loop = loads + WMMA only.
__global__ __launch_bounds__(256)
void gemm_readout_bf(const __bf16* __restrict__ Xbf, const __bf16* __restrict__ Wbf,
                     const float* __restrict__ bias, float* __restrict__ out) {
    const int lane    = threadIdx.x & 31;
    const int wave    = threadIdx.x >> 5;
    const int laneLow = lane & 15;
    const int hi      = lane >> 4;

    const int nTile = blockIdx.x * 8 + wave;       // 0..1999
    const int mBase = blockIdx.y * 64;             // 4 m-tiles of 16 rows
    const int col   = nTile * 16 + laneLow;

    const __bf16* __restrict__ wrow = Wbf + (size_t)col * K_;

    v8f acc[4] = {};

    for (int k = 0; k < K_; k += 32) {
        // B fragment (32x16 bf16): lane holds col, K = hi*16+0..15 contiguous.
        v16bf bfrag = *reinterpret_cast<const v16bf*>(wrow + k + hi * 16);
#pragma unroll
        for (int mi = 0; mi < 4; ++mi) {
            v16bf afrag = load_afrag(Xbf, mBase + mi * 16 + laneLow, k, hi);
            acc[mi] = __builtin_amdgcn_wmma_f32_16x16x32_bf16(
                false, afrag, false, bfrag, (short)0, acc[mi], false, false);
        }
    }

    const float bv = bias[col];
#pragma unroll
    for (int mi = 0; mi < 4; ++mi) {
#pragma unroll
        for (int r = 0; r < 8; ++r) {
            int m = mBase + mi * 16 + hi * 8 + r;
            out[(size_t)m * V_ + col] = acc[mi][r] + bv;
        }
    }
}

// Fallback (workspace too small for bf16 W copy): convert W inline.
__global__ __launch_bounds__(256)
void gemm_readout_mixed(const __bf16* __restrict__ Xbf, const float* __restrict__ W,
                        const float* __restrict__ bias, float* __restrict__ out) {
    const int lane    = threadIdx.x & 31;
    const int wave    = threadIdx.x >> 5;
    const int laneLow = lane & 15;
    const int hi      = lane >> 4;

    const int nTile = blockIdx.x * 8 + wave;
    const int mBase = blockIdx.y * 64;
    const int col   = nTile * 16 + laneLow;

    const float* __restrict__ wrow = W + (size_t)col * K_;

    v8f acc[4] = {};

    for (int k = 0; k < K_; k += 32) {
        const float4* wp = reinterpret_cast<const float4*>(wrow + k + hi * 16);
        float4 w0 = wp[0], w1 = wp[1], w2 = wp[2], w3 = wp[3];
        v16bf bfrag;
        cvt8(bfrag, 0, w0, w1);
        cvt8(bfrag, 8, w2, w3);
#pragma unroll
        for (int mi = 0; mi < 4; ++mi) {
            v16bf afrag = load_afrag(Xbf, mBase + mi * 16 + laneLow, k, hi);
            acc[mi] = __builtin_amdgcn_wmma_f32_16x16x32_bf16(
                false, afrag, false, bfrag, (short)0, acc[mi], false, false);
        }
    }

    const float bv = bias[col];
#pragma unroll
    for (int mi = 0; mi < 4; ++mi) {
#pragma unroll
        for (int r = 0; r < 8; ++r) {
            int m = mBase + mi * 16 + hi * 8 + r;
            out[(size_t)m * V_ + col] = acc[mi][r] + bv;
        }
    }
}

// ---------------------------------------------------------------------------
extern "C" void kernel_launch(void* const* d_in, const int* in_sizes, int n_in,
                              void* d_out, int out_size, void* d_ws, size_t ws_size,
                              hipStream_t stream) {
    (void)in_sizes; (void)n_in; (void)out_size;

    const int*   idx  = (const int*)  d_in[0];
    const int*   edge = (const int*)  d_in[1];   // [2, E]
    const float* emb  = (const float*)d_in[2];
    const float* Gx   = (const float*)d_in[3];
    const float* Gy   = (const float*)d_in[4];
    const float* Gs   = (const float*)d_in[5];
    const float* W    = (const float*)d_in[6];
    const float* b_ro = (const float*)d_in[7];

    float* out = (float*)d_out;
    float* ws  = (float*)d_ws;

    // fp32 state region
    float* sigma = ws;                    // E
    float* x     = sigma + E_;            // B*N
    float* y     = x + B_ * N_;           // B*N
    float* A     = y + B_ * N_;           // B*N
    float* ynew  = A + B_ * N_;           // B*N
    float* xacc  = ynew + B_ * N_;        // B*N
    const size_t f32Floats = (size_t)E_ + 5u * B_ * N_;   // 147456

    // bf16 regions
    unsigned short* Xbf = (unsigned short*)(ws + f32Floats);          // 512*2048
    unsigned short* Wbf = Xbf + (size_t)B_ * T_ * K_;                 // 32000*2048

    const size_t needBase = f32Floats * 4 + (size_t)B_ * T_ * K_ * 2;
    const size_t needFull = needBase + (size_t)V_ * K_ * 2;
    const bool fullPath = (ws_size >= needFull);

    const int* srcI = edge;
    const int* dstI = edge + E_;

    const int threads = 256;
    const int eBlocks = E_ / threads;          // 256
    const int sBlocks = (B_ * N_) / threads;   // 64

    // sigma0 = 0
    zero_f<<<eBlocks, threads, 0, stream>>>(sigma, E_);

    // Pre-convert W_ro to bf16 (streamed once; L2-resident afterwards).
    if (fullPath) {
        const int nvec = (V_ * K_) / 8;        // 8,192,000
        cvt_f32_bf16_vec8<<<nvec / threads, threads, 0, stream>>>(
            W, (uint4*)Wbf, nvec);
    }

    for (int t = 0; t < T_; ++t) {
        gather_x<<<sBlocks, threads, 0, stream>>>(emb, idx, t, x, y);
        for (int layer = 0; layer < 2; ++layer) {
            zero3<<<sBlocks, threads, 0, stream>>>(A, ynew, xacc, B_ * N_);
            edge_pass1<<<eBlocks, threads, 0, stream>>>(srcI, dstI, Gs, sigma, x, y, A);
            edge_pass2<<<eBlocks, threads, 0, stream>>>(srcI, dstI, Gy, A, ynew);
            edge_pass3<<<eBlocks, threads, 0, stream>>>(srcI, dstI, Gx, ynew, xacc);
            finalize_layer<<<sBlocks, threads, 0, stream>>>(ynew, xacc, x, y, Xbf, t);
        }
    }

    // Deferred batched readout: logits = X @ W^T + b
    dim3 gg(2000 / 8, 512 / 64);   // (250, 8)
    if (fullPath) {
        gemm_readout_bf<<<gg, 256, 0, stream>>>((const __bf16*)Xbf, (const __bf16*)Wbf,
                                                b_ro, out);
    } else {
        gemm_readout_mixed<<<gg, 256, 0, stream>>>((const __bf16*)Xbf, W, b_ro, out);
    }

    // sigma appended after logits
    copy_f<<<eBlocks, threads, 0, stream>>>(out + (size_t)B_ * T_ * V_, sigma, E_);
}
